// OpenAD_DGCNN_61735859912963
// MI455X (gfx1250) — matrix operand order repaired and checked by reading, not compiled, not run on hardware
//
#include <hip/hip_runtime.h>
#include <hip/hip_bf16.h>
#include <stdint.h>

typedef __attribute__((ext_vector_type(16))) _Float16 v16h;
typedef __attribute__((ext_vector_type(8)))  float    v8f;

#define BB   8
#define NN   2048
#define KNN  40
#define L1   (BB*NN*KNN)   /* 655360 rows for edge-conv GEMMs */
#define L2   (BB*NN)       /* 16384 points */
#define NCLS 19
#define TDIM 512

__device__ __forceinline__ float lrelu(float z){ return z >= 0.f ? z : 0.2f*z; }

static inline unsigned gdiv(size_t n, int b){ return (unsigned)((n + (size_t)b - 1)/(size_t)b); }

// ---------------- layout transforms ----------------

// (B,3,N) channel-major -> (B*N, 3) point-major
__global__ void __launch_bounds__(256) to_pm_kernel(const float* __restrict__ x, float* __restrict__ out)
{
    int p = blockIdx.x*blockDim.x + threadIdx.x;
    if (p >= L2) return;
    int b = p / NN, n = p % NN;
    for (int c = 0; c < 3; ++c)
        out[(size_t)p*3 + c] = x[(size_t)b*3*NN + (size_t)c*NN + n];
}

__global__ void __launch_bounds__(256) norms_kernel(const float* __restrict__ X, float* __restrict__ nrm, int C)
{
    int p = blockIdx.x*blockDim.x + threadIdx.x;
    if (p >= L2) return;
    float s = 0.f;
    for (int c = 0; c < C; ++c){ float v = X[(size_t)p*C + c]; s += v*v; }
    nrm[p] = s;
}

// top-K (largest neg-dist) per point, per batch
__global__ void __launch_bounds__(128) knn_kernel(const float* __restrict__ X, const float* __restrict__ nrm,
                                                  int* __restrict__ idx, int C)
{
    int p = blockIdx.x*blockDim.x + threadIdx.x;
    if (p >= L2) return;
    int b = p / NN;
    float q[64];
    for (int c = 0; c < C; ++c) q[c] = X[(size_t)p*C + c];
    float bestd[KNN]; int besti[KNN];
    for (int i = 0; i < KNN; ++i){ bestd[i] = -3.4e38f; besti[i] = 0; }
    const float* Xb = X + (size_t)b*NN*C;
    const float* Nb = nrm + (size_t)b*NN;
    float nq = nrm[p];
    for (int m = 0; m < NN; ++m){
        float dot = 0.f;
        for (int c = 0; c < C; ++c) dot += q[c]*Xb[(size_t)m*C + c];
        float nd = 2.f*dot - nq - Nb[m];
        if (nd > bestd[KNN-1]){
            int j = KNN-1;
            while (j > 0 && bestd[j-1] < nd){ bestd[j] = bestd[j-1]; besti[j] = besti[j-1]; --j; }
            bestd[j] = nd; besti[j] = m;
        }
    }
    for (int i = 0; i < KNN; ++i) idx[(size_t)p*KNN + i] = besti[i];
}

// edge features: row (p,k) = [x_nb - x_c , x_c]  -> ROWS (L1 x 2C)
__global__ void __launch_bounds__(256) edge_rows_kernel(const float* __restrict__ X, const int* __restrict__ idx,
                                                        float* __restrict__ R, int C)
{
    size_t rr = (size_t)blockIdx.x*blockDim.x + threadIdx.x;
    if (rr >= (size_t)L1) return;
    size_t p = rr / KNN;
    int b = (int)(p / NN);
    int nb = idx[rr];
    const float* xc = X + p*C;
    const float* xn = X + ((size_t)b*NN + nb)*C;
    float* out = R + rr*(2*C);
    for (int c = 0; c < C; ++c){ float cv = xc[c]; out[c] = xn[c] - cv; out[C + c] = cv; }
}

// ---------------- WMMA packing (CDNA5 16x16x32 f16 fragments) ----------------
// A fragment (16x32, ISA 7.12.2): lane = m + 16*hi; vgpr v holds K pair
//   K = (v<4 ? 2v : 16+2(v-4)) + 8*hi + half
// Optionally fuses the producer's BatchNorm (scale/shift) + LeakyReLU into the pack.
__global__ void __launch_bounds__(256) pack_a_kernel(const float* __restrict__ R, _Float16* __restrict__ A,
                                                     int L, int C, int ltiles, int kcs,
                                                     const float* __restrict__ scale,
                                                     const float* __restrict__ shift)
{
    size_t tid = (size_t)blockIdx.x*blockDim.x + threadIdx.x;
    size_t total = (size_t)ltiles*kcs*32;
    if (tid >= total) return;
    int lane = (int)(tid & 31);
    size_t frag = tid >> 5;
    int kcid = (int)(frag % kcs);
    int lt   = (int)(frag / kcs);
    int m = lane & 15, hi = lane >> 4;
    int row = lt*16 + m;
    _Float16* dst = A + frag*512 + (size_t)lane*16;
    for (int e = 0; e < 16; ++e){
        int v = e >> 1, h = e & 1;
        int kk = (v < 4 ? 2*v : 16 + 2*(v-4)) + 8*hi + h;
        int col = kcid*32 + kk;
        float val = 0.f;
        if (row < L && col < C){
            val = R[(size_t)row*C + col];
            if (scale) val = lrelu(val*scale[col] + shift[col]);
        }
        dst[e] = (_Float16)val;
    }
}

// B fragment (32x16): lane -> n = lane%16, element e -> k (same table order as A). B[k,n] = W[n_oc, k]
__global__ void __launch_bounds__(256) pack_b_kernel(const float* __restrict__ W, _Float16* __restrict__ Bp,
                                                     int OC, int C, int octiles, int kcs)
{
    size_t tid = (size_t)blockIdx.x*blockDim.x + threadIdx.x;
    size_t total = (size_t)octiles*kcs*32;
    if (tid >= total) return;
    int lane = (int)(tid & 31);
    size_t frag = tid >> 5;
    int kcid = (int)(frag % kcs);
    int ot   = (int)(frag / kcs);
    int n = lane & 15, hi = lane >> 4;
    _Float16* dst = Bp + frag*512 + (size_t)lane*16;
    for (int e = 0; e < 16; ++e){
        int k = kcid*32 + (e < 8 ? 8*hi + e : 16 + 8*hi + (e - 8));
        int oc = ot*16 + n;
        float val = (oc < OC && k < C) ? W[(size_t)oc*C + k] : 0.f;
        dst[e] = (_Float16)val;
    }
}

// ---------------- WMMA GEMM: Out[L x OC] = A(LxC) * W^T ----------------
// Each wave: one 16-row L tile x FOUR consecutive 16-col OC tiles (OC % 64 == 0 everywhere),
// reusing every A fragment 4x -> 4x less HBM read traffic on the streamed operand.
__global__ void __launch_bounds__(256) wmma_gemm_kernel(const _Float16* __restrict__ A,
                                                        const _Float16* __restrict__ Bp,
                                                        float* __restrict__ Y,
                                                        int ltiles, int octiles, int kcs)
{
    int ogroups = octiles >> 2;
    size_t wave = ((size_t)blockIdx.x*blockDim.x + threadIdx.x) >> 5;
    int lane = threadIdx.x & 31;
    size_t nwaves = (size_t)ltiles*ogroups;
    if (wave >= nwaves) return;
    int lt = (int)(wave / ogroups);
    int og = (int)(wave % ogroups);
    const v16h* Af = (const v16h*)A  + (size_t)lt*kcs*32 + lane;
    const v16h* Bf = (const v16h*)Bp + (size_t)(og*4)*kcs*32 + lane;
    v8f acc0 = {}, acc1 = {}, acc2 = {}, acc3 = {};
    for (int i = 0; i < kcs; ++i){
        if (i + 1 < kcs) __builtin_prefetch(Af + (size_t)(i+1)*32, 0, 3);   // global_prefetch_b8
        v16h a  = Af[(size_t)i*32];
        v16h b0 = Bf[((size_t)0*kcs + i)*32];
        v16h b1 = Bf[((size_t)1*kcs + i)*32];
        v16h b2 = Bf[((size_t)2*kcs + i)*32];
        v16h b3 = Bf[((size_t)3*kcs + i)*32];
        acc0 = __builtin_amdgcn_wmma_f32_16x16x32_f16(false, a, false, b0, (short)0, acc0, false, false);
        acc1 = __builtin_amdgcn_wmma_f32_16x16x32_f16(false, a, false, b1, (short)0, acc1, false, false);
        acc2 = __builtin_amdgcn_wmma_f32_16x16x32_f16(false, a, false, b2, (short)0, acc2, false, false);
        acc3 = __builtin_amdgcn_wmma_f32_16x16x32_f16(false, a, false, b3, (short)0, acc3, false, false);
    }
    int OC = octiles*16;
    int n = lane & 15, hi = lane >> 4;
    float* Yr = Y + (size_t)lt*16*OC + og*64 + n;
    #pragma unroll
    for (int v = 0; v < 8; ++v){
        int m = v + 8*hi;                 // D layout: vgpr v, lane -> (M,N) per ISA 7.12.2
        float* yr = Yr + (size_t)m*OC;
        yr[ 0] = acc0[v];
        yr[16] = acc1[v];
        yr[32] = acc2[v];
        yr[48] = acc3[v];
    }
}

// ---------------- BatchNorm (batch statistics), deterministic ----------------
__global__ void __launch_bounds__(256) col_stats_kernel(const float* __restrict__ Y, float* __restrict__ sum,
                                                        float* __restrict__ sq, int L, int OC)
{
    int c = blockIdx.x;
    float s = 0.f, q = 0.f;
    for (int r = threadIdx.x; r < L; r += 256){
        float v = Y[(size_t)r*OC + c]; s += v; q += v*v;
    }
    __shared__ float ss[256], qq[256];
    ss[threadIdx.x] = s; qq[threadIdx.x] = q; __syncthreads();
    for (int o = 128; o > 0; o >>= 1){
        if ((int)threadIdx.x < o){ ss[threadIdx.x] += ss[threadIdx.x+o]; qq[threadIdx.x] += qq[threadIdx.x+o]; }
        __syncthreads();
    }
    if (threadIdx.x == 0){ sum[c] = ss[0]; sq[c] = qq[0]; }
}

__global__ void __launch_bounds__(256) bn_finalize_kernel(const float* __restrict__ sum, const float* __restrict__ sq,
                                                          const float* __restrict__ g, const float* __restrict__ bt,
                                                          float* __restrict__ scale, float* __restrict__ shift,
                                                          int OC, float invL)
{
    int c = blockIdx.x*blockDim.x + threadIdx.x;
    if (c >= OC) return;
    float m = sum[c]*invL;
    float v = sq[c]*invL - m*m;
    float rs = rsqrtf(v + 1e-5f);
    float sc = g[c]*rs;
    scale[c] = sc; shift[c] = bt[c] - m*sc;
}

// ---------------- pooling (optionally fused with producer BN + LeakyReLU) ----------------
__global__ void __launch_bounds__(256) maxk_kernel(const float* __restrict__ R, float* __restrict__ X, int C,
                                                   const float* __restrict__ scale, const float* __restrict__ shift)
{
    size_t i = (size_t)blockIdx.x*blockDim.x + threadIdx.x;   // i = p*C + c
    if (i >= (size_t)L2*C) return;
    size_t p = i / C; int c = (int)(i % C);
    float sc = scale ? scale[c] : 1.f;
    float sh = shift ? shift[c] : 0.f;
    const float* base = R + (p*KNN)*(size_t)C + c;
    float m = -3.4e38f;
    for (int k = 0; k < KNN; ++k){
        float v = base[(size_t)k*C];
        if (scale) v = lrelu(v*sc + sh);
        m = fmaxf(m, v);
    }
    X[i] = m;
}

__global__ void __launch_bounds__(256) maxn_kernel(const float* __restrict__ R, float* __restrict__ G, int C,
                                                   const float* __restrict__ scale, const float* __restrict__ shift)
{
    int i = blockIdx.x*blockDim.x + threadIdx.x;              // i = b*C + c
    if (i >= BB*C) return;
    int b = i / C, c = i % C;
    float sc = scale ? scale[c] : 1.f;
    float sh = shift ? shift[c] : 0.f;
    const float* base = R + ((size_t)b*NN)*C + c;
    float m = -3.4e38f;
    for (int n = 0; n < NN; ++n){
        float v = base[(size_t)n*C];
        if (scale) v = lrelu(v*sc + sh);
        m = fmaxf(m, v);
    }
    G[i] = m;
}

// ---------------- tiny FC path (transform net head) ----------------
__global__ void __launch_bounds__(256) fc_kernel(const float* __restrict__ in, const float* __restrict__ W,
                                                 const float* __restrict__ bias, float* __restrict__ out,
                                                 int R, int Cin, int OC)
{
    int i = blockIdx.x*blockDim.x + threadIdx.x;
    if (i >= R*OC) return;
    int r = i / OC, o = i % OC;
    float s = bias ? bias[o] : 0.f;
    const float* x = in + (size_t)r*Cin;
    const float* w = W + (size_t)o*Cin;
    for (int c = 0; c < Cin; ++c) s += x[c]*w[c];
    out[i] = s;
}

__global__ void __launch_bounds__(256) bn_batch_lrelu_kernel(const float* __restrict__ in, const float* __restrict__ g,
                                                             const float* __restrict__ bt, float* __restrict__ out, int C)
{
    int c = blockIdx.x*blockDim.x + threadIdx.x;
    if (c >= C) return;
    float s = 0.f, q = 0.f;
    for (int r = 0; r < BB; ++r){ float v = in[(size_t)r*C + c]; s += v; q += v*v; }
    float m = s/(float)BB, var = q/(float)BB - m*m, rs = rsqrtf(var + 1e-5f);
    for (int r = 0; r < BB; ++r){
        float z = (in[(size_t)r*C + c] - m)*rs*g[c] + bt[c];
        out[(size_t)r*C + c] = lrelu(z);
    }
}

// x'[b,:,n] = x[b,:,n]^T @ t[b] -> point-major (L2 x 3)
__global__ void __launch_bounds__(256) transform_apply_kernel(const float* __restrict__ xin, const float* __restrict__ T,
                                                              float* __restrict__ out)
{
    int i = blockIdx.x*blockDim.x + threadIdx.x;              // i = p*3 + d
    if (i >= L2*3) return;
    int p = i / 3, d = i % 3;
    int b = p / NN, n = p % NN;
    float s = 0.f;
    for (int c = 0; c < 3; ++c)
        s += xin[(size_t)b*3*NN + (size_t)c*NN + n] * T[b*9 + c*3 + d];
    out[i] = s;
}

// ---------------- concat builders ----------------
__global__ void __launch_bounds__(256) concat3_kernel(const float* __restrict__ x1, const float* __restrict__ x2,
                                                      const float* __restrict__ x3, float* __restrict__ R)
{
    size_t i = (size_t)blockIdx.x*blockDim.x + threadIdx.x;   // i = p*192 + j
    if (i >= (size_t)L2*192) return;
    size_t p = i / 192; int j = (int)(i % 192);
    float v = (j < 64) ? x1[p*64 + j] : (j < 128) ? x2[p*64 + (j-64)] : x3[p*64 + (j-128)];
    R[i] = v;
}

__global__ void __launch_bounds__(256) concat_head_kernel(const float* __restrict__ g, const float* __restrict__ x1,
                                                          const float* __restrict__ x2, const float* __restrict__ x3,
                                                          float* __restrict__ R)
{
    size_t i = (size_t)blockIdx.x*blockDim.x + threadIdx.x;   // i = p*1216 + j
    if (i >= (size_t)L2*1216) return;
    size_t p = i / 1216; int j = (int)(i % 1216);
    int b = (int)(p / NN);
    float v;
    if (j < 1024)      v = g[(size_t)b*1024 + j];
    else if (j < 1088) v = x1[p*64 + (j-1024)];
    else if (j < 1152) v = x2[p*64 + (j-1088)];
    else               v = x3[p*64 + (j-1152)];
    R[i] = v;
}

// ---------------- cosine-similarity head ----------------
__global__ void __launch_bounds__(256) tn_norm_kernel(const float* __restrict__ tf, float* __restrict__ TN)
{
    int cls = blockIdx.x*blockDim.x + threadIdx.x;
    if (cls >= NCLS) return;
    float s = 0.f;
    for (int d = 0; d < TDIM; ++d){ float v = tf[(size_t)d*NCLS + cls]; s += v*v; }
    float rs = rsqrtf(s + 1e-12f);
    for (int d = 0; d < TDIM; ++d) TN[(size_t)d*NCLS + cls] = tf[(size_t)d*NCLS + cls]*rs;
}

__global__ void __launch_bounds__(256) pnorm_kernel(const float* __restrict__ H, float* __restrict__ pn)
{
    int p = blockIdx.x*blockDim.x + threadIdx.x;
    if (p >= L2) return;
    float s = 0.f;
    for (int d = 0; d < TDIM; ++d){ float v = H[(size_t)p*TDIM + d]; s += v*v; }
    pn[p] = rsqrtf(s + 1e-12f);
}

__global__ void __launch_bounds__(256) logits_kernel(const float* __restrict__ H, const float* __restrict__ TN,
                                                     const float* __restrict__ pn, float* __restrict__ out)
{
    int i = blockIdx.x*blockDim.x + threadIdx.x;              // i = p*NCLS + cls
    if (i >= L2*NCLS) return;
    int p = i / NCLS, cls = i % NCLS;
    int b = p / NN, n = p % NN;
    float s = 0.f;
    for (int d = 0; d < TDIM; ++d) s += H[(size_t)p*TDIM + d]*TN[(size_t)d*NCLS + cls];
    out[((size_t)b*NCLS + cls)*NN + n] = s*pn[p]*(1.0f/0.07f);
}

// ---------------- host orchestration ----------------
struct ConvBufs { _Float16* APACK; _Float16* BPACK; float* SUM; float* SQ; float* SCALE; float* SHIFT; };

// pack (optionally applying previous layer's BN+lrelu) -> WMMA GEMM -> (optional) stats+finalize.
// On return with bn=true, cb.SCALE/cb.SHIFT hold THIS layer's fold, to be consumed by the next stage.
static void conv_gemm(hipStream_t s, const float* rows_in, int L, int Cin,
                      const float* in_scale, const float* in_shift,
                      const float* W, int OC, const float* g, const float* bt,
                      bool bn, float* out, const ConvBufs& cb)
{
    int kcs = (Cin + 31)/32;
    int ltiles = L/16;
    int octiles = OC/16;
    pack_a_kernel<<<gdiv((size_t)ltiles*kcs*32,256),256,0,s>>>(rows_in, cb.APACK, L, Cin, ltiles, kcs,
                                                               in_scale, in_shift);
    pack_b_kernel<<<gdiv((size_t)octiles*kcs*32,256),256,0,s>>>(W, cb.BPACK, OC, Cin, octiles, kcs);
    wmma_gemm_kernel<<<gdiv((size_t)ltiles*(octiles>>2)*32,256),256,0,s>>>(cb.APACK, cb.BPACK, out,
                                                                           ltiles, octiles, kcs);
    if (bn){
        col_stats_kernel<<<OC,256,0,s>>>(out, cb.SUM, cb.SQ, L, OC);
        bn_finalize_kernel<<<gdiv(OC,256),256,0,s>>>(cb.SUM, cb.SQ, g, bt, cb.SCALE, cb.SHIFT, OC, 1.0f/(float)L);
    }
}

extern "C" void kernel_launch(void* const* d_in, const int* in_sizes, int n_in,
                              void* d_out, int out_size, void* d_ws, size_t ws_size,
                              hipStream_t stream)
{
    (void)in_sizes; (void)n_in; (void)out_size; (void)ws_size;
    // input order (setup_inputs dict order, params flattened in insertion order)
    const float* x_in  = (const float*)d_in[0];
    const float* textf = (const float*)d_in[1];
    const float* t_c1w = (const float*)d_in[2];
    const float* t_c2w = (const float*)d_in[3];
    const float* t_c3w = (const float*)d_in[4];
    const float* t_l1w = (const float*)d_in[5];
    const float* t_l2w = (const float*)d_in[6];
    const float* c1w   = (const float*)d_in[7];
    const float* c2w   = (const float*)d_in[8];
    const float* c3w   = (const float*)d_in[9];
    const float* c4w   = (const float*)d_in[10];
    const float* c5w   = (const float*)d_in[11];
    const float* c6w   = (const float*)d_in[12];
    const float* c7w   = (const float*)d_in[13];
    const float* c8w   = (const float*)d_in[14];
    const float* c9w   = (const float*)d_in[15];
    const float* t_b1g = (const float*)d_in[16]; const float* t_b1b = (const float*)d_in[17];
    const float* t_b2g = (const float*)d_in[18]; const float* t_b2b = (const float*)d_in[19];
    const float* t_b3g = (const float*)d_in[20]; const float* t_b3b = (const float*)d_in[21];
    const float* t_l1g = (const float*)d_in[22]; const float* t_l1b = (const float*)d_in[23];
    const float* t_l2g = (const float*)d_in[24]; const float* t_l2b = (const float*)d_in[25];
    const float* b1g = (const float*)d_in[26]; const float* b1b = (const float*)d_in[27];
    const float* b2g = (const float*)d_in[28]; const float* b2b = (const float*)d_in[29];
    const float* b3g = (const float*)d_in[30]; const float* b3b = (const float*)d_in[31];
    const float* b4g = (const float*)d_in[32]; const float* b4b = (const float*)d_in[33];
    const float* b5g = (const float*)d_in[34]; const float* b5b = (const float*)d_in[35];
    const float* b6g = (const float*)d_in[36]; const float* b6b = (const float*)d_in[37];
    const float* b7g = (const float*)d_in[38]; const float* b7b = (const float*)d_in[39];
    const float* b8g = (const float*)d_in[40]; const float* b8b = (const float*)d_in[41];
    const float* t_tw = (const float*)d_in[42];
    const float* t_tb = (const float*)d_in[43];
    float* out = (float*)d_out;

    // ---- workspace layout (bump allocate, 256B aligned) ----
    char* base = (char*)d_ws; size_t off = 0;
    auto alloc = [&](size_t bytes) -> char* {
        char* p = base + off; off += (bytes + 255) & ~(size_t)255; return p;
    };
    int*      IDX    = (int*)     alloc((size_t)L1*4);
    float*    XPMraw = (float*)   alloc((size_t)L2*3*4);
    float*    XPM0   = (float*)   alloc((size_t)L2*3*4);
    float*    X1     = (float*)   alloc((size_t)L2*64*4);
    float*    X2     = (float*)   alloc((size_t)L2*64*4);
    float*    X3     = (float*)   alloc((size_t)L2*64*4);
    float*    XT     = (float*)   alloc((size_t)L2*128*4);
    float*    NRM    = (float*)   alloc((size_t)L2*4);
    float*    ROWS   = (float*)   alloc((size_t)L1*128*4);   // shared edge/GEMM-out buffer
    _Float16* APACK  = (_Float16*)alloc((size_t)L1*128*2);
    _Float16* BPACK  = (_Float16*)alloc((size_t)1216*1024*2);
    float*    SUM    = (float*)   alloc(1024*4);
    float*    SQ     = (float*)   alloc(1024*4);
    float*    SCALE  = (float*)   alloc(1024*4);
    float*    SHIFT  = (float*)   alloc(1024*4);
    float*    TG     = (float*)   alloc((size_t)BB*1024*4);
    float*    FCA    = (float*)   alloc((size_t)BB*1024*4);
    float*    FCB    = (float*)   alloc((size_t)BB*1024*4);
    float*    TMAT   = (float*)   alloc((size_t)BB*9*4);
    float*    PN     = (float*)   alloc((size_t)L2*4);
    float*    TN     = (float*)   alloc((size_t)TDIM*NCLS*4);
    ConvBufs cb = { APACK, BPACK, SUM, SQ, SCALE, SHIFT };

    // ---- transform net on raw x ----
    to_pm_kernel<<<gdiv(L2,256),256,0,stream>>>(x_in, XPMraw);
    norms_kernel<<<gdiv(L2,256),256,0,stream>>>(XPMraw, NRM, 3);
    knn_kernel<<<gdiv(L2,128),128,0,stream>>>(XPMraw, NRM, IDX, 3);
    edge_rows_kernel<<<gdiv(L1,256),256,0,stream>>>(XPMraw, IDX, ROWS, 3);
    conv_gemm(stream, ROWS, L1, 6,   nullptr, nullptr, t_c1w, 64,   t_b1g, t_b1b, true, ROWS, cb);
    conv_gemm(stream, ROWS, L1, 64,  SCALE,   SHIFT,   t_c2w, 128,  t_b2g, t_b2b, true, ROWS, cb);
    maxk_kernel<<<gdiv((size_t)L2*128,256),256,0,stream>>>(ROWS, XT, 128, SCALE, SHIFT);
    conv_gemm(stream, XT,   L2, 128, nullptr, nullptr, t_c3w, 1024, t_b3g, t_b3b, true, ROWS, cb);
    maxn_kernel<<<gdiv(BB*1024,256),256,0,stream>>>(ROWS, TG, 1024, SCALE, SHIFT);
    fc_kernel<<<gdiv(BB*512,256),256,0,stream>>>(TG,  t_l1w, nullptr, FCA, BB, 1024, 512);
    bn_batch_lrelu_kernel<<<gdiv(512,256),256,0,stream>>>(FCA, t_l1g, t_l1b, FCB, 512);
    fc_kernel<<<gdiv(BB*256,256),256,0,stream>>>(FCB, t_l2w, nullptr, FCA, BB, 512, 256);
    bn_batch_lrelu_kernel<<<gdiv(256,256),256,0,stream>>>(FCA, t_l2g, t_l2b, FCB, 256);
    fc_kernel<<<gdiv(BB*9,256),256,0,stream>>>(FCB, t_tw, t_tb, TMAT, BB, 256, 9);
    transform_apply_kernel<<<gdiv(L2*3,256),256,0,stream>>>(x_in, TMAT, XPM0);

    // ---- EdgeConv stage 1 (aligned points) ----
    norms_kernel<<<gdiv(L2,256),256,0,stream>>>(XPM0, NRM, 3);
    knn_kernel<<<gdiv(L2,128),128,0,stream>>>(XPM0, NRM, IDX, 3);
    edge_rows_kernel<<<gdiv(L1,256),256,0,stream>>>(XPM0, IDX, ROWS, 3);
    conv_gemm(stream, ROWS, L1, 6,  nullptr, nullptr, c1w, 64, b1g, b1b, true, ROWS, cb);
    conv_gemm(stream, ROWS, L1, 64, SCALE,   SHIFT,   c2w, 64, b2g, b2b, true, ROWS, cb);
    maxk_kernel<<<gdiv((size_t)L2*64,256),256,0,stream>>>(ROWS, X1, 64, SCALE, SHIFT);

    // ---- EdgeConv stage 2 ----
    norms_kernel<<<gdiv(L2,256),256,0,stream>>>(X1, NRM, 64);
    knn_kernel<<<gdiv(L2,128),128,0,stream>>>(X1, NRM, IDX, 64);
    edge_rows_kernel<<<gdiv(L1,256),256,0,stream>>>(X1, IDX, ROWS, 64);
    conv_gemm(stream, ROWS, L1, 128, nullptr, nullptr, c3w, 64, b3g, b3b, true, ROWS, cb);
    conv_gemm(stream, ROWS, L1, 64,  SCALE,   SHIFT,   c4w, 64, b4g, b4b, true, ROWS, cb);
    maxk_kernel<<<gdiv((size_t)L2*64,256),256,0,stream>>>(ROWS, X2, 64, SCALE, SHIFT);

    // ---- EdgeConv stage 3 ----
    norms_kernel<<<gdiv(L2,256),256,0,stream>>>(X2, NRM, 64);
    knn_kernel<<<gdiv(L2,128),128,0,stream>>>(X2, NRM, IDX, 64);
    edge_rows_kernel<<<gdiv(L1,256),256,0,stream>>>(X2, IDX, ROWS, 64);
    conv_gemm(stream, ROWS, L1, 128, nullptr, nullptr, c5w, 64, b5g, b5b, true, ROWS, cb);
    maxk_kernel<<<gdiv((size_t)L2*64,256),256,0,stream>>>(ROWS, X3, 64, SCALE, SHIFT);

    // ---- global feature ----
    concat3_kernel<<<gdiv((size_t)L2*192,256),256,0,stream>>>(X1, X2, X3, ROWS);
    conv_gemm(stream, ROWS, L2, 192, nullptr, nullptr, c6w, 1024, b6g, b6b, true, ROWS, cb);
    maxn_kernel<<<gdiv(BB*1024,256),256,0,stream>>>(ROWS, TG, 1024, SCALE, SHIFT);

    // ---- per-point head ----
    concat_head_kernel<<<gdiv((size_t)L2*1216,256),256,0,stream>>>(TG, X1, X2, X3, ROWS);
    conv_gemm(stream, ROWS, L2, 1216, nullptr, nullptr, c7w, 512, b7g, b7b, true,  ROWS, cb);
    conv_gemm(stream, ROWS, L2, 512,  SCALE,   SHIFT,   c8w, 256, b8g, b8b, true,  ROWS, cb);
    conv_gemm(stream, ROWS, L2, 256,  SCALE,   SHIFT,   c9w, 512, nullptr, nullptr, false, ROWS, cb);

    // ---- cosine similarity logits ----
    tn_norm_kernel<<<1,32,0,stream>>>(textf, TN);
    pnorm_kernel<<<gdiv(L2,256),256,0,stream>>>(ROWS, PN);
    logits_kernel<<<gdiv(L2*NCLS,256),256,0,stream>>>(ROWS, TN, PN, out);
}
